// hklayer_srh_31301721653360
// MI455X (gfx1250) — compile-verified, matrix operand order checked by reading, not run on hardware
//
#include <hip/hip_runtime.h>
#include <cstdint>

#define BB 32
#define CC 512
#define CMID 128
#define HH 2048

typedef _Float16 v16h __attribute__((ext_vector_type(16)));
typedef _Float16 v8h  __attribute__((ext_vector_type(8)));
typedef _Float16 v4h  __attribute__((ext_vector_type(4)));
typedef float    v8f  __attribute__((ext_vector_type(8)));
typedef unsigned int v4u __attribute__((ext_vector_type(4)));
typedef int      v8i  __attribute__((ext_vector_type(8)));
typedef int      v4i  __attribute__((ext_vector_type(4)));

__device__ __forceinline__ v8f wmma16(v16h a, v16h b, v8f c) {
  return __builtin_amdgcn_wmma_f32_16x16x32_f16(false, a, false, b, (short)0, c, false, false);
}

// Load one 16x32 f16 fragment (A-layout) from LDS rows of stride 40 halves.
// lane<16: row=row0+lane, K halves [0..8) and [16..24); lane>=16: K [8..16),[24..32)
__device__ __forceinline__ v16h lds_frag(const _Float16* base, int row0, int lane) {
  int r  = row0 + (lane & 15);
  int o8 = (lane >> 4) << 3;
  const _Float16* p = base + r * 40;
  v8h lo = *(const v8h*)(p + o8);
  v8h hi = *(const v8h*)(p + 16 + o8);
  v16h out;
#pragma unroll
  for (int i = 0; i < 8; ++i) { out[i] = lo[i]; out[i + 8] = hi[i]; }
  return out;
}

// ---------------- small prep kernels ----------------

__global__ void init_ws(float* st) {
  for (int i = threadIdx.x; i < 3200; i += 256) st[i] = 0.f;
}

__global__ void reduce_w(const float* __restrict__ W, float* wsum) {
  float l[4] = {0.f, 0.f, 0.f, 0.f};
  for (int i = blockIdx.x * 256 + threadIdx.x; i < CMID * CMID * 9; i += 128 * 256) {
    int k = i % 9;
    int cls = (k >= 3 && k <= 5) ? 0 : ((k == 2 || k == 6) ? 1 : ((k == 1 || k == 7) ? 2 : 3));
    l[cls] += W[i];
  }
  __shared__ float sh[4];
  if (threadIdx.x < 4) sh[threadIdx.x] = 0.f;
  __syncthreads();
#pragma unroll
  for (int c = 0; c < 4; ++c) atomicAdd(&sh[c], l[c]);
  __syncthreads();
  if (threadIdx.x < 4) atomicAdd(&wsum[threadIdx.x], sh[threadIdx.x]);
}

__global__ void prep_coef(const float* __restrict__ wsum, float* coef) {
  if (threadIdx.x == 0) {
    const float cnt = (float)(CMID * CMID);
    float a[4];
    a[0] = wsum[0] / (3.f * cnt + 1e-10f);
    a[1] = wsum[1] / (2.f * cnt + 1e-10f);
    a[2] = wsum[2] / (2.f * cnt + 1e-10f);
    a[3] = wsum[3] / (2.f * cnt + 1e-10f);
    float m = fmaxf(fmaxf(a[0], a[1]), fmaxf(a[2], a[3]));
    float s = 0.f;
#pragma unroll
    for (int i = 0; i < 4; ++i) { a[i] = __expf(a[i] - m); s += a[i]; }
#pragma unroll
    for (int i = 0; i < 4; ++i) a[i] /= s;
    for (int k = 0; k < 9; ++k) {
      float c = a[3];
      if (k >= 1 && k <= 7) c += a[2];
      if (k >= 2 && k <= 6) c += a[1];
      if (k >= 3 && k <= 5) c += a[0];
      coef[k] = c;
    }
  }
}

// wk[k][o][i] = f16(W[o,i,k] * coef[k])   (k-major so each GEMM slice is contiguous)
__global__ void scale_w(const float* __restrict__ W, const float* __restrict__ coef,
                        _Float16* __restrict__ wk) {
  int idx = blockIdx.x * 256 + threadIdx.x;
  if (idx >= 9 * CMID * CMID) return;
  int k = idx / (CMID * CMID);
  int r = idx - k * (CMID * CMID);   // r = o*128 + i
  wk[idx] = (_Float16)(W[r * 9 + k] * coef[k]);
}

__global__ void finalize_bn(const float* __restrict__ gs, const float* __restrict__ gq,
                            const float* __restrict__ g, const float* __restrict__ be,
                            float* sc, float* sh, int n, float invc) {
  int i = blockIdx.x * blockDim.x + threadIdx.x;
  if (i < n) {
    float mean = gs[i] * invc;
    float var  = gq[i] * invc - mean * mean;
    float s    = g[i] * rsqrtf(var + 1e-5f);
    sc[i] = s;
    sh[i] = be[i] - mean * s;
  }
}

// ---------------- GEMM1: h1 = conv1_w (128x512) * x[b] (512x2048) + bias, + BN1 stats ----------------

__global__ __launch_bounds__(256) void gemm1_kernel(
    const float* __restrict__ x, const float* __restrict__ w1, const float* __restrict__ b1,
    _Float16* __restrict__ h1, float* __restrict__ gsum, float* __restrict__ gsq) {
  __shared__ __align__(16) _Float16 As[128 * 40];
  __shared__ __align__(16) _Float16 Bs[128 * 40];
  __shared__ float ssum[128], ssq[128];
  const int t = threadIdx.x, lane = t & 31, wv = t >> 5;
  const int b = blockIdx.y, h0 = blockIdx.x * 128;
  const int mw = (wv & 3) * 32, nw = (wv >> 2) * 64;
  if (t < 128) { ssum[t] = 0.f; ssq[t] = 0.f; }
  const float* xb = x + (size_t)b * CC * HH;
  v8f acc[2][4] = {};
  for (int kb = 0; kb < 16; ++kb) {
    __syncthreads();
#pragma unroll 4
    for (int i = t; i < 4096; i += 256) {        // A: 128x32 of conv1_w
      int r = i >> 5, c = i & 31;
      As[r * 40 + c] = (_Float16)w1[r * CC + kb * 32 + c];
    }
#pragma unroll 4
    for (int i = t; i < 4096; i += 256) {        // B: 32x128 of x, transposed (N-major) in LDS
      int kk = i >> 7, n = i & 127;
      Bs[n * 40 + kk] = (_Float16)xb[(size_t)(kb * 32 + kk) * HH + h0 + n];
    }
    if (kb < 15)
      __builtin_prefetch(&xb[(size_t)(kb * 32 + 32 + lane) * HH + h0 + (wv << 4)], 0, 1);
    __syncthreads();
    v16h af0 = lds_frag(As, mw, lane);
    v16h af1 = lds_frag(As, mw + 16, lane);
#pragma unroll
    for (int ni = 0; ni < 4; ++ni) {
      v16h bf = lds_frag(Bs, nw + ni * 16, lane);
      acc[0][ni] = wmma16(af0, bf, acc[0][ni]);
      acc[1][ni] = wmma16(af1, bf, acc[1][ni]);
    }
  }
  const int hi = lane >> 4, nl = lane & 15;
#pragma unroll
  for (int mi = 0; mi < 2; ++mi)
#pragma unroll
    for (int i = 0; i < 8; ++i) {
      int M = mw + mi * 16 + hi * 8 + i;
      float bias = b1[M];
      float ps = 0.f, pq = 0.f;
#pragma unroll
      for (int ni = 0; ni < 4; ++ni) {
        float v = acc[mi][ni][i] + bias;
        int hg = h0 + nw + ni * 16 + nl;
        h1[((size_t)b * CMID + M) * HH + hg] = (_Float16)v;
        ps += v; pq += v * v;
      }
      atomicAdd(&ssum[M], ps);
      atomicAdd(&ssq[M], pq);
    }
  __syncthreads();
  if (t < 128) { atomicAdd(&gsum[t], ssum[t]); atomicAdd(&gsq[t], ssq[t]); }
}

// ---------------- conv9 as 9x accumulated GEMM; BN1 folded into B staging; TDM loads A ----------------

__global__ __launch_bounds__(256) void conv9_kernel(
    const _Float16* __restrict__ h1, const _Float16* __restrict__ wk,
    const float* __restrict__ s1, const float* __restrict__ t1,
    _Float16* __restrict__ y, float* __restrict__ gsum, float* __restrict__ gsq) {
  __shared__ __align__(16) _Float16 As[128 * 40];
  __shared__ __align__(16) _Float16 Bs[128 * 40];
  __shared__ float ssum[128], ssq[128], shs[128], sht[128];
  const int t = threadIdx.x, lane = t & 31, wv = t >> 5;
  const int b = blockIdx.y, h0 = blockIdx.x * 128;
  const int mw = (wv & 3) * 32, nw = (wv >> 2) * 64;
  if (t < 128) { ssum[t] = 0.f; ssq[t] = 0.f; shs[t] = s1[t]; sht[t] = t1[t]; }
  v8f acc[2][4] = {};
  for (int k9 = 0; k9 < 9; ++k9)
    for (int kc = 0; kc < 4; ++kc) {
      __syncthreads();
      if (wv == 0) {
        // TDM: load 128-row x 32-half f16 tile of wk[k9][:, kc*32..+32] into As with
        // 16B padding per 64B row (LDS row stride 40 halves).
        unsigned ldsoff = (unsigned)(uintptr_t)&As[0];
        unsigned long long ga =
            (unsigned long long)(uintptr_t)(wk + (size_t)k9 * CMID * CMID + kc * 32);
        v4u g0;
        g0[0] = 1u;                                   // count=1, user descriptor
        g0[1] = ldsoff;                               // lds_addr
        g0[2] = (unsigned)ga;                         // global_addr[31:0]
        g0[3] = (unsigned)(ga >> 32) | (2u << 30);    // global_addr[56:32] | type=2
        v8i g1;
        g1[0] = (1 << 16) | (1 << 20) | (3 << 22) | (3 << 25); // 2B elems, pad: 16dw interval, 4dw amount
        g1[1] = (128 << 16);                          // tensor_dim0 = 128
        g1[2] = (128 << 16);                          // tensor_dim1 = 128
        g1[3] = (32 << 16);                           // tile_dim0 = 32
        g1[4] = 128;                                  // tile_dim1 = 128, tile_dim2 = 0
        g1[5] = 128;                                  // tensor_dim0_stride = 128
        g1[6] = 0; g1[7] = 0;                         // tensor_dim1_stride = 0
        v4i z4 = {0, 0, 0, 0};
        v8i z8 = {0, 0, 0, 0, 0, 0, 0, 0};
        __builtin_amdgcn_tensor_load_to_lds(g0, g1, z4, z4, z8, 0);
        __builtin_amdgcn_s_wait_tensorcnt(0);
      }
#pragma unroll 4
      for (int i = t; i < 4096; i += 256) {         // B: shifted, BN1-normalized h1 slab
        int r = i >> 7, n = i & 127;
        int ch = kc * 32 + r;
        int hs = h0 + n + k9 - 4;
        float v = 0.f;
        if (hs >= 0 && hs < HH)
          v = shs[ch] * (float)h1[((size_t)b * CMID + ch) * HH + hs] + sht[ch];
        Bs[n * 40 + r] = (_Float16)v;
      }
      __syncthreads();
      v16h af0 = lds_frag(As, mw, lane);
      v16h af1 = lds_frag(As, mw + 16, lane);
#pragma unroll
      for (int ni = 0; ni < 4; ++ni) {
        v16h bf = lds_frag(Bs, nw + ni * 16, lane);
        acc[0][ni] = wmma16(af0, bf, acc[0][ni]);
        acc[1][ni] = wmma16(af1, bf, acc[1][ni]);
      }
    }
  const int hi = lane >> 4, nl = lane & 15;
#pragma unroll
  for (int mi = 0; mi < 2; ++mi)
#pragma unroll
    for (int i = 0; i < 8; ++i) {
      int M = mw + mi * 16 + hi * 8 + i;
      float ps = 0.f, pq = 0.f;
#pragma unroll
      for (int ni = 0; ni < 4; ++ni) {
        float v = acc[mi][ni][i];
        int hg = h0 + nw + ni * 16 + nl;
        y[((size_t)b * CMID + M) * HH + hg] = (_Float16)v;
        ps += v; pq += v * v;
      }
      atomicAdd(&ssum[M], ps);
      atomicAdd(&ssq[M], pq);
    }
  __syncthreads();
  if (t < 128) { atomicAdd(&gsum[t], ssum[t]); atomicAdd(&gsq[t], ssq[t]); }
}

// ---------------- GEMM2: z = conv2_w (512x128) * relu(BNs(y)) + bias, + BN2 stats ----------------

__global__ __launch_bounds__(256) void gemm2_kernel(
    const _Float16* __restrict__ y, const float* __restrict__ w2, const float* __restrict__ b2,
    const float* __restrict__ ss, const float* __restrict__ ts,
    _Float16* __restrict__ z, float* __restrict__ gsum, float* __restrict__ gsq) {
  __shared__ __align__(16) _Float16 As[128 * 40];
  __shared__ __align__(16) _Float16 Bs[128 * 40];
  __shared__ float ssum[128], ssq[128], shs[128], sht[128];
  const int t = threadIdx.x, lane = t & 31, wv = t >> 5;
  const int b = blockIdx.z, mbase = blockIdx.y * 128, h0 = blockIdx.x * 128;
  const int mw = (wv & 3) * 32, nw = (wv >> 2) * 64;
  if (t < 128) { ssum[t] = 0.f; ssq[t] = 0.f; shs[t] = ss[t]; sht[t] = ts[t]; }
  v8f acc[2][4] = {};
  for (int kb = 0; kb < 4; ++kb) {
    __syncthreads();
#pragma unroll 4
    for (int i = t; i < 4096; i += 256) {
      int r = i >> 5, c = i & 31;
      As[r * 40 + c] = (_Float16)w2[(size_t)(mbase + r) * CMID + kb * 32 + c];
    }
#pragma unroll 4
    for (int i = t; i < 4096; i += 256) {
      int r = i >> 7, n = i & 127;
      int ch = kb * 32 + r;
      float v = shs[ch] * (float)y[((size_t)b * CMID + ch) * HH + h0 + n] + sht[ch];
      Bs[n * 40 + r] = (_Float16)fmaxf(v, 0.f);
    }
    if (kb < 3)
      __builtin_prefetch(&y[((size_t)b * CMID + kb * 32 + 32 + lane) * HH + h0 + (wv << 4)], 0, 1);
    __syncthreads();
    v16h af0 = lds_frag(As, mw, lane);
    v16h af1 = lds_frag(As, mw + 16, lane);
#pragma unroll
    for (int ni = 0; ni < 4; ++ni) {
      v16h bf = lds_frag(Bs, nw + ni * 16, lane);
      acc[0][ni] = wmma16(af0, bf, acc[0][ni]);
      acc[1][ni] = wmma16(af1, bf, acc[1][ni]);
    }
  }
  const int hi = lane >> 4, nl = lane & 15;
#pragma unroll
  for (int mi = 0; mi < 2; ++mi)
#pragma unroll
    for (int i = 0; i < 8; ++i) {
      int M = mw + mi * 16 + hi * 8 + i;
      float bias = b2[mbase + M];
      float ps = 0.f, pq = 0.f;
#pragma unroll
      for (int ni = 0; ni < 4; ++ni) {
        float v = acc[mi][ni][i] + bias;
        int hg = h0 + nw + ni * 16 + nl;
        z[((size_t)b * CC + mbase + M) * HH + hg] = (_Float16)v;
        ps += v; pq += v * v;
      }
      atomicAdd(&ssum[M], ps);
      atomicAdd(&ssq[M], pq);
    }
  __syncthreads();
  if (t < 128) { atomicAdd(&gsum[mbase + t], ssum[t]); atomicAdd(&gsq[mbase + t], ssq[t]); }
}

// ---------------- epilogue: out = relu(BN2(z)) + x ----------------

__global__ __launch_bounds__(256) void final_kernel(
    const _Float16* __restrict__ z, const float* __restrict__ x,
    const float* __restrict__ s2, const float* __restrict__ t2, float* __restrict__ out) {
  int tid = blockIdx.x * 256 + threadIdx.x;       // one float4 / half4 per thread
  int idx4 = tid * 4;
  int c = (idx4 >> 11) & 511;                     // H = 2048
  float sc = s2[c], sh = t2[c];
  v4h zz = ((const v4h*)z)[tid];
  float4 xx = ((const float4*)x)[tid];
  float4 o;
  o.x = fmaxf(sc * (float)zz[0] + sh, 0.f) + xx.x;
  o.y = fmaxf(sc * (float)zz[1] + sh, 0.f) + xx.y;
  o.z = fmaxf(sc * (float)zz[2] + sh, 0.f) + xx.z;
  o.w = fmaxf(sc * (float)zz[3] + sh, 0.f) + xx.w;
  ((float4*)out)[tid] = o;
}

// ---------------- host ----------------

extern "C" void kernel_launch(void* const* d_in, const int* in_sizes, int n_in,
                              void* d_out, int out_size, void* d_ws, size_t ws_size,
                              hipStream_t stream) {
  const float* x    = (const float*)d_in[0];
  const float* W    = (const float*)d_in[1];
  const float* c1w  = (const float*)d_in[2];
  const float* c1b  = (const float*)d_in[3];
  const float* c2w  = (const float*)d_in[4];
  const float* c2b  = (const float*)d_in[5];
  const float* bn1g = (const float*)d_in[6];
  const float* bn1b = (const float*)d_in[7];
  const float* bnsg = (const float*)d_in[8];
  const float* bnsb = (const float*)d_in[9];
  const float* bn2g = (const float*)d_in[10];
  const float* bn2b = (const float*)d_in[11];

  float* st   = (float*)d_ws;
  float* wsum = st + 0;     // 4
  float* coef = st + 4;     // 9
  float* b1s  = st + 16;    float* b1q = st + 144;   // 128 each
  float* bss  = st + 272;   float* bsq = st + 400;   // 128 each
  float* b2s  = st + 528;   float* b2q = st + 1040;  // 512 each
  float* s1   = st + 1552;  float* t1  = st + 1680;
  float* ssv  = st + 1808;  float* tsv = st + 1936;
  float* s2   = st + 2064;  float* t2  = st + 2576;

  _Float16* wk = (_Float16*)((char*)d_ws + 16384);
  _Float16* h1 = (_Float16*)((char*)d_ws + (1u << 20));
  _Float16* yb = (_Float16*)((char*)d_ws + 18874368u);
  _Float16* zb = (_Float16*)((char*)d_ws + 35651584u);
  float* out = (float*)d_out;

  const float invc = 1.f / (float)(BB * HH);

  init_ws<<<1, 256, 0, stream>>>(st);
  reduce_w<<<128, 256, 0, stream>>>(W, wsum);
  prep_coef<<<1, 32, 0, stream>>>(wsum, coef);
  scale_w<<<576, 256, 0, stream>>>(W, coef, wk);

  gemm1_kernel<<<dim3(16, 32), 256, 0, stream>>>(x, c1w, c1b, h1, b1s, b1q);
  finalize_bn<<<1, 128, 0, stream>>>(b1s, b1q, bn1g, bn1b, s1, t1, 128, invc);

  conv9_kernel<<<dim3(16, 32), 256, 0, stream>>>(h1, wk, s1, t1, yb, bss, bsq);
  finalize_bn<<<1, 128, 0, stream>>>(bss, bsq, bnsg, bnsb, ssv, tsv, 128, invc);

  gemm2_kernel<<<dim3(16, 4, 32), 256, 0, stream>>>(yb, c2w, c2b, ssv, tsv, zb, b2s, b2q);
  finalize_bn<<<2, 256, 0, stream>>>(b2s, b2q, bn2g, bn2b, s2, t2, 512, invc);

  final_kernel<<<32768, 256, 0, stream>>>(zb, x, s2, t2, out);
}
